// MultiLatentAttention_45019847197306
// MI455X (gfx1250) — matrix operand order repaired and checked by reading, not compiled, not run on hardware
//
#include <hip/hip_runtime.h>
#include <hip/hip_bf16.h>

// ---------------------------------------------------------------------------
// MLA forward for MI455X (gfx1250, wave32, WMMA f32<-f16 16x16x32).
// All matmuls are NT-GEMMs C = alpha * A(MxK) * W(NxK)^T + bias + add, run
// through one LDS-staged double-buffered WMMA kernel.  A-tiles stream in via
// per-lane global_load_async_to_lds_b128 (ASYNCcnt); B-tiles via one TDM
// tensor_load_to_lds descriptor per stage (TENSORcnt), with TDM padding
// reproducing the bank-conflict-free 80-B LDS row pitch.
// ---------------------------------------------------------------------------

typedef _Float16 half_t;
typedef __attribute__((ext_vector_type(16))) _Float16 v16h;
typedef __attribute__((ext_vector_type(8)))  _Float16 v8h;
typedef __attribute__((ext_vector_type(8)))  float    v8f;
typedef __attribute__((ext_vector_type(4)))  unsigned v4u;
typedef __attribute__((ext_vector_type(8)))  unsigned v8u;

namespace cfg {
constexpr int S    = 2048;   // sequence length
constexpr int HID  = 4096;   // hidden
constexpr int H    = 32;     // heads
constexpr int QL   = 1536;   // q low-rank
constexpr int KVL  = 512;    // kv low-rank
constexpr int RH   = 2048;   // rope hidden (q)
constexpr int DC   = 128;    // content head dim
constexpr int DR   = 64;     // rope head dim
constexpr int DH   = 192;    // full head dim
constexpr float EPS = 1.1920929e-07f; // finfo(float32).eps
}

// ---------------------------------------------------------------------------
// CDNA5 async global->LDS copy (ASYNCcnt), CDNA5 ISA
// ---------------------------------------------------------------------------
__device__ __forceinline__ void async_copy_b128(unsigned lds_off, const void* gptr) {
  asm volatile("global_load_async_to_lds_b128 %0, %1, off"
               :: "v"(lds_off), "v"(gptr)
               : "memory");
}
__device__ __forceinline__ void wait_asynccnt0() {
  asm volatile("s_wait_asynccnt 0x0" ::: "memory");
}

// ---------------------------------------------------------------------------
// CDNA5 Tensor Data Mover: one 2-D tile (rows x 32 halves, row stride
// ld_halves) DMA'd from global into LDS at lds_off, with 16 B of LDS padding
// appended after every 64-B row (pad_interval=3 -> 16 DWORDs, pad_amount=3 ->
// 4 DWORDs).  D# per CDNA5 ISA; groups 2/3 NULL
// (2-D tensor).  Tracked by TENSORcnt.
// ---------------------------------------------------------------------------
__device__ __forceinline__ void tdm_load_tile_2d(unsigned lds_off, const void* gptr,
                                                 unsigned rows, unsigned ld_halves) {
  const unsigned long long ga = (unsigned long long)(size_t)gptr;
  v4u g0;
  g0[0] = 1u;                                              // count=1, user D#
  g0[1] = lds_off;                                         // lds_addr (bytes)
  g0[2] = (unsigned)(ga & 0xFFFFFFFFu);                    // global_addr[31:0]
  g0[3] = (unsigned)((ga >> 32) & 0x01FFFFFFu) | (2u << 30); // addr[56:32]|type=2
  v8u g1;
  g1[0] = (1u << 16)        // data_size = 2 bytes
        | (1u << 20)        // pad_enable
        | (3u << 22)        // pad_interval: 16 DWORDs (one 32-half row)
        | (3u << 25);       // pad_amount: 4 DWORDs (8 halves)
  g1[1] = 32u << 16;        // tensor_dim0 = 32 (bits 79:48, low half)
  g1[2] = rows << 16;       // tensor_dim0 hi = 0 | tensor_dim1 = rows (low half)
  g1[3] = 32u << 16;        // tensor_dim1 hi = 0 | tile_dim0 = 32
  g1[4] = rows;             // tile_dim1 = rows | tile_dim2 = 0
  g1[5] = ld_halves;        // tensor_dim0_stride[31:0] (data_size units)
  g1[6] = 0u;               // tensor_dim0_stride[47:32] | tensor_dim1_stride lo
  g1[7] = 0u;               // tensor_dim1_stride hi
  asm volatile("tensor_load_to_lds %0, %1" :: "s"(g0), "s"(g1) : "memory");
}

// ---------------------------------------------------------------------------
// f32 -> f16 conversion (grid-stride)
// ---------------------------------------------------------------------------
__global__ __launch_bounds__(256) void cvt_f32_to_f16(const float* __restrict__ src,
                                                      half_t* __restrict__ dst,
                                                      size_t n) {
  size_t i = (size_t)blockIdx.x * blockDim.x + threadIdx.x;
  size_t stride = (size_t)gridDim.x * blockDim.x;
  for (; i < n; i += stride) dst[i] = (half_t)src[i];
}

// ---------------------------------------------------------------------------
// Tiled NT-GEMM:  C[M,N] = alpha * A[M,K] * B[N,K]^T (+bias[n]) (+add[m,n])
// Block tile 256x64x32, 256 threads = 8 waves (4 along M x 2 along N),
// wave tile 64x32 = 4x2 WMMA 16x16x32 fragments.  K-tiles double-buffered in
// LDS: A via per-lane async-b128, B via one TDM descriptor (wave 0).
// Rows padded to 40 halves (80 B) for conflict-free ds_load_b128.
// Requires M % 256 == 0, N % 64 == 0, K % 32 == 0 (true for all shapes here).
// ---------------------------------------------------------------------------
template <bool HB, bool HM, bool WF, bool WH>
__global__ __launch_bounds__(256) void gemm_nt_wmma(
    const half_t* __restrict__ A, int lda,
    const half_t* __restrict__ B, int ldb,
    const float*  __restrict__ bias,
    const float*  __restrict__ addm, int ldadd,
    float alpha,
    float*  __restrict__ Cf,
    half_t* __restrict__ Ch, int ldc,
    int K) {
  constexpr int BM = 256, BN = 64, BK = 32, PK = 40;  // PK: padded row (halves)
  __shared__ half_t sA[2][BM * PK];
  __shared__ half_t sB[2][BN * PK];

  const int tid  = threadIdx.x;
  const int lane = tid & 31;
  const int wave = tid >> 5;
  const int lr   = lane & 15;   // row/col within 16-tile
  const int hi   = lane >> 4;   // half-wave selector
  const int wm   = wave & 3;    // 4 waves along M, 64 rows each
  const int wn   = wave >> 2;   // 2 waves along N, 32 cols each
  const int mBase = blockIdx.y * BM;
  const int nBase = blockIdx.x * BN;
  const int aoff = hi * 16;  // A frag: lanes 0-15 take K 0..15, lanes 16-31 take K 16..31
  const int boff = hi * 16;  // B frag: lanes 0-15 take K 0..15, lanes 16-31 take K 16..31

  // Issue one K-stage: A tile by per-lane async copies, B tile by TDM.
  auto stage = [&](int buf, int kk) {
    const half_t* Ag = A + (size_t)mBase * lda + kk;
    half_t* la = sA[buf];
#pragma unroll
    for (int r = 0; r < 4; ++r) {                    // A: 256 rows x 4 chunks
      const int c   = tid + 256 * r;
      const int row = c >> 2;
      const int ch  = (c & 3) * 8;                   // halves
      async_copy_b128((unsigned)(size_t)&la[row * PK + ch],
                      Ag + (size_t)row * lda + ch);
    }
    if (wave == 0)                                   // B: one TDM 2-D tile DMA
      tdm_load_tile_2d((unsigned)(size_t)&sB[buf][0],
                       B + (size_t)nBase * ldb + kk, BN, (unsigned)ldb);
  };
  auto stage_wait = [&]() {
    wait_asynccnt0();
    __builtin_amdgcn_s_wait_tensorcnt(0);
  };

  v8f acc[4][2] = {};

  stage(0, 0);
  stage_wait();
  __syncthreads();

  int cur = 0;
  for (int kk = 0; kk < K; kk += BK) {
    const bool more = (kk + BK) < K;
    if (more) stage(cur ^ 1, kk + BK);   // overlap next-stage DMA with WMMA

    const half_t* la = sA[cur];
    const half_t* lb = sB[cur];
    v16h a[4], b[2];
#pragma unroll
    for (int i = 0; i < 4; ++i) {
      const half_t* base = la + (wm * 64 + i * 16 + lr) * PK;
      const v8h lo = *(const v8h*)(base + aoff);
      const v8h hv = *(const v8h*)(base + 8 + aoff);
#pragma unroll
      for (int t = 0; t < 8; ++t) { a[i][t] = lo[t]; a[i][t + 8] = hv[t]; }
    }
#pragma unroll
    for (int j = 0; j < 2; ++j)
      b[j] = *(const v16h*)(lb + (wn * 32 + j * 16 + lr) * PK + boff);

#pragma unroll
    for (int i = 0; i < 4; ++i)
#pragma unroll
      for (int j = 0; j < 2; ++j)
        acc[i][j] = __builtin_amdgcn_wmma_f32_16x16x32_f16(
            false, a[i], false, b[j], (short)0, acc[i][j], false, false);

    if (more) stage_wait();
    __syncthreads();
    cur ^= 1;
  }

  // C/D layout: lane holds column (lane%16), rows v + 8*(lane/16)
#pragma unroll
  for (int i = 0; i < 4; ++i) {
    const int mrb = mBase + wm * 64 + i * 16 + 8 * hi;
#pragma unroll
    for (int j = 0; j < 2; ++j) {
      const int nc = nBase + wn * 32 + j * 16 + lr;
      const float bval = HB ? bias[nc] : 0.0f;
#pragma unroll
      for (int v = 0; v < 8; ++v) {
        const int mr = mrb + v;
        float val = alpha * acc[i][j][v] + bval;
        if (HM) val += addm[(size_t)mr * ldadd + nc];
        if (WF) Cf[(size_t)mr * ldc + nc] = val;
        if (WH) Ch[(size_t)mr * ldc + nc] = (half_t)val;
      }
    }
  }
}

// ---------------------------------------------------------------------------
// Row softmax: P[row,:] = softmax(Sc[row,:]) as f16.  One block per row.
// ---------------------------------------------------------------------------
__global__ __launch_bounds__(256) void softmax_row_kernel(const float* __restrict__ Sc,
                                                          half_t* __restrict__ P,
                                                          int n) {
  const int row = blockIdx.x;
  const float* src = Sc + (size_t)row * n;
  half_t* dst = P + (size_t)row * n;
  __shared__ float red[8];
  const int wid = threadIdx.x >> 5;

  float m = -3.402823466e38f;
  for (int i = threadIdx.x; i < n; i += blockDim.x) m = fmaxf(m, src[i]);
#pragma unroll
  for (int off = 16; off > 0; off >>= 1) m = fmaxf(m, __shfl_xor(m, off, 32));
  if ((threadIdx.x & 31) == 0) red[wid] = m;
  __syncthreads();
  if (threadIdx.x == 0) {
    float t = red[0];
    for (int i = 1; i < 8; ++i) t = fmaxf(t, red[i]);
    red[0] = t;
  }
  __syncthreads();
  m = red[0];
  __syncthreads();

  float s = 0.0f;
  for (int i = threadIdx.x; i < n; i += blockDim.x) s += expf(src[i] - m);
#pragma unroll
  for (int off = 16; off > 0; off >>= 1) s += __shfl_xor(s, off, 32);
  if ((threadIdx.x & 31) == 0) red[wid] = s;
  __syncthreads();
  if (threadIdx.x == 0) {
    float t = 0.0f;
    for (int i = 0; i < 8; ++i) t += red[i];
    red[0] = t;
  }
  __syncthreads();
  const float inv = 1.0f / red[0];

  for (int i = threadIdx.x; i < n; i += blockDim.x)
    dst[i] = (half_t)(expf(src[i] - m) * inv);
}

// ---------------------------------------------------------------------------
// Assemble Q: concat(q_content[s, h*128+d], rope(q_rope_pre)[s, h*64+j]),
// RMS-norm over 192, write f16 as Q[h][s][d].  Grid (S, H), block 192.
// RoPE is over the full 2048-dim vector (rotate-half, D/2 = 1024).
// ---------------------------------------------------------------------------
__global__ __launch_bounds__(192) void build_q_kernel(const float* __restrict__ qc,
                                                      const float* __restrict__ qr,
                                                      half_t* __restrict__ Q) {
  using namespace cfg;
  const int s = blockIdx.x, h = blockIdx.y, d = threadIdx.x;
  float val;
  if (d < DC) {
    val = qc[(size_t)s * HID + h * DC + d];
  } else {
    const int j = h * DR + (d - DC);          // 0..2047
    const float* row = qr + (size_t)s * RH;
    const int hd = RH / 2;                    // 1024
    if (j < hd) {
      const float fr  = powf(10000.0f, -(float)j / (float)hd);
      const float ang = (float)s * fr;
      val = row[j] * cosf(ang) - row[j + hd] * sinf(ang);
    } else {
      const int jj = j - hd;
      const float fr  = powf(10000.0f, -(float)jj / (float)hd);
      const float ang = (float)s * fr;
      val = row[j] * cosf(ang) + row[jj] * sinf(ang);
    }
  }
  float ss = val * val;
#pragma unroll
  for (int off = 16; off > 0; off >>= 1) ss += __shfl_xor(ss, off, 32);
  __shared__ float red[6];
  if ((threadIdx.x & 31) == 0) red[threadIdx.x >> 5] = ss;
  __syncthreads();
  const float tot = red[0] + red[1] + red[2] + red[3] + red[4] + red[5];
  const float rms = rsqrtf(tot / (float)DH + EPS);
  Q[((size_t)h * S + s) * DH + d] = (half_t)(val * rms);
}

// Same for K: rope part is a 64-dim vector (rotate-half, D/2 = 32),
// broadcast to all heads before the norm.
__global__ __launch_bounds__(192) void build_k_kernel(const float* __restrict__ kc,
                                                      const float* __restrict__ kr,
                                                      half_t* __restrict__ Kh) {
  using namespace cfg;
  const int s = blockIdx.x, h = blockIdx.y, d = threadIdx.x;
  float val;
  if (d < DC) {
    val = kc[(size_t)s * HID + h * DC + d];
  } else {
    const int j = d - DC;                     // 0..63
    const float* row = kr + (size_t)s * DR;
    const int hd = DR / 2;                    // 32
    if (j < hd) {
      const float fr  = powf(10000.0f, -(float)j / (float)hd);
      const float ang = (float)s * fr;
      val = row[j] * cosf(ang) - row[j + hd] * sinf(ang);
    } else {
      const int jj = j - hd;
      const float fr  = powf(10000.0f, -(float)jj / (float)hd);
      const float ang = (float)s * fr;
      val = row[j] * cosf(ang) + row[jj] * sinf(ang);
    }
  }
  float ss = val * val;
#pragma unroll
  for (int off = 16; off > 0; off >>= 1) ss += __shfl_xor(ss, off, 32);
  __shared__ float red[6];
  if ((threadIdx.x & 31) == 0) red[threadIdx.x >> 5] = ss;
  __syncthreads();
  const float tot = red[0] + red[1] + red[2] + red[3] + red[4] + red[5];
  const float rms = rsqrtf(tot / (float)DH + EPS);
  Kh[((size_t)h * S + s) * DH + d] = (half_t)(val * rms);
}

// Vt[h][n][s] = v[s, h*128 + n]  (f32 -> f16, transposed so O = P * Vt^T is NT)
__global__ __launch_bounds__(256) void build_v_kernel(const float* __restrict__ v,
                                                      half_t* __restrict__ Vt) {
  using namespace cfg;
  const size_t total = (size_t)H * DC * S;
  size_t i = (size_t)blockIdx.x * blockDim.x + threadIdx.x;
  if (i >= total) return;
  const size_t s = i % S;
  const size_t n = (i / S) % DC;
  const size_t h = i / ((size_t)S * DC);
  Vt[i] = (half_t)v[s * HID + h * DC + n];
}

// ---------------------------------------------------------------------------
// Host-side orchestration
// ---------------------------------------------------------------------------
extern "C" void kernel_launch(void* const* d_in, const int* in_sizes, int n_in,
                              void* d_out, int out_size, void* d_ws, size_t ws_size,
                              hipStream_t stream) {
  (void)in_sizes; (void)n_in; (void)out_size;
  using namespace cfg;

  const float* hidden  = (const float*)d_in[0];
  const float* mask    = (const float*)d_in[1];
  const float* Wq_lr   = (const float*)d_in[2];
  const float* bq_lr   = (const float*)d_in[3];
  const float* Wq_rope = (const float*)d_in[4];
  const float* bq_rope = (const float*)d_in[5];
  const float* Wq_c    = (const float*)d_in[6];
  const float* bq_c    = (const float*)d_in[7];
  const float* Wkv     = (const float*)d_in[8];
  const float* bkv     = (const float*)d_in[9];
  const float* Wk_rope = (const float*)d_in[10];
  const float* bk_rope = (const float*)d_in[11];
  const float* Wk_c    = (const float*)d_in[12];
  const float* bk_c    = (const float*)d_in[13];
  const float* Wv      = (const float*)d_in[14];
  const float* bv      = (const float*)d_in[15];
  const float* Wo      = (const float*)d_in[16];
  const float* bo      = (const float*)d_in[17];
  float* out = (float*)d_out;

  // ---- workspace carve-out (256-B aligned slabs) ----
  char* p = (char*)d_ws;
  auto alloc = [&](size_t bytes) -> char* {
    char* r = p;
    p += (bytes + 255) & ~(size_t)255;
    return r;
  };
  half_t* Xh    = (half_t*)alloc((size_t)S * HID * 2);
  half_t* Wqlrh = (half_t*)alloc((size_t)QL * HID * 2);
  half_t* Wqrph = (half_t*)alloc((size_t)RH * QL * 2);
  half_t* Wqch  = (half_t*)alloc((size_t)HID * QL * 2);
  half_t* Wkvh  = (half_t*)alloc((size_t)KVL * HID * 2);
  half_t* Wkrph = (half_t*)alloc((size_t)DR * HID * 2);
  half_t* Wkch  = (half_t*)alloc((size_t)HID * KVL * 2);
  half_t* Wvh   = (half_t*)alloc((size_t)HID * KVL * 2);
  half_t* Woh   = (half_t*)alloc((size_t)HID * HID * 2);
  half_t* cqh   = (half_t*)alloc((size_t)S * QL * 2);
  half_t* ckvh  = (half_t*)alloc((size_t)S * KVL * 2);
  half_t* Qh    = (half_t*)alloc((size_t)H * S * DH * 2);
  half_t* Kh    = (half_t*)alloc((size_t)H * S * DH * 2);
  half_t* Vt    = (half_t*)alloc((size_t)H * DC * S * 2);
  half_t* Pmat  = (half_t*)alloc((size_t)S * S * 2);
  half_t* Oh    = (half_t*)alloc((size_t)S * HID * 2);   // attn output, f16
  float*  qcf   = (float*)alloc((size_t)S * HID * 4);
  float*  qrf   = (float*)alloc((size_t)S * RH * 4);
  float*  kcf   = (float*)alloc((size_t)S * HID * 4);
  float*  krf   = (float*)alloc((size_t)S * DR * 4);
  float*  vf    = (float*)alloc((size_t)S * HID * 4);
  float*  scf   = (float*)alloc((size_t)S * S * 4);
  if ((size_t)(p - (char*)d_ws) > ws_size) return;  // not enough scratch

  auto cvt = [&](const float* src, half_t* dst, size_t n) {
    int blocks = (int)((n + 255) / 256);
    if (blocks > 65535) blocks = 65535;
    cvt_f32_to_f16<<<blocks, 256, 0, stream>>>(src, dst, n);
  };
  // Dispatch on epilogue configuration (compile-time specialized kernels).
  auto gemm = [&](const half_t* A, int lda, const half_t* B, int ldb,
                  const float* bias, const float* addm, int ldadd, float alpha,
                  float* Cf, half_t* Ch, int ldc, int M, int N, int K) {
    dim3 g(N / 64, M / 256);
#define GEMM_GO(HB, HM, WF, WH)                                              \
    gemm_nt_wmma<HB, HM, WF, WH><<<g, 256, 0, stream>>>(                     \
        A, lda, B, ldb, bias, addm, ldadd, alpha, Cf, Ch, ldc, K)
    if (bias) {
      if (Cf) GEMM_GO(true, false, true, false);
      else    GEMM_GO(true, false, false, true);
    } else if (addm) {
      GEMM_GO(false, true, true, false);
    } else {
      GEMM_GO(false, false, false, true);
    }
#undef GEMM_GO
  };

  // ---- f16 copies of activations + weights ----
  cvt(hidden,  Xh,    (size_t)S * HID);
  cvt(Wq_lr,   Wqlrh, (size_t)QL * HID);
  cvt(Wq_rope, Wqrph, (size_t)RH * QL);
  cvt(Wq_c,    Wqch,  (size_t)HID * QL);
  cvt(Wkv,     Wkvh,  (size_t)KVL * HID);
  cvt(Wk_rope, Wkrph, (size_t)DR * HID);
  cvt(Wk_c,    Wkch,  (size_t)HID * KVL);
  cvt(Wv,      Wvh,   (size_t)HID * KVL);
  cvt(Wo,      Woh,   (size_t)HID * HID);

  // ---- projection GEMMs ----
  gemm(Xh,  HID, Wqlrh, HID, bq_lr,   nullptr, 0, 1.0f, nullptr, cqh, QL,  S, QL,  HID); // c_q (f16)
  gemm(cqh, QL,  Wqch,  QL,  bq_c,    nullptr, 0, 1.0f, qcf, nullptr, HID, S, HID, QL);  // q_content (f32)
  gemm(cqh, QL,  Wqrph, QL,  bq_rope, nullptr, 0, 1.0f, qrf, nullptr, RH,  S, RH,  QL);  // q_rope pre (f32)
  gemm(Xh,  HID, Wkvh,  HID, bkv,     nullptr, 0, 1.0f, nullptr, ckvh, KVL, S, KVL, HID);// c_kv (f16)
  gemm(ckvh, KVL, Wkch, KVL, bk_c,    nullptr, 0, 1.0f, kcf, nullptr, HID, S, HID, KVL); // k_content (f32)
  gemm(Xh,  HID, Wkrph, HID, bk_rope, nullptr, 0, 1.0f, krf, nullptr, DR,  S, DR,  HID); // k_rope pre (f32)
  gemm(ckvh, KVL, Wvh,  KVL, bv,      nullptr, 0, 1.0f, vf,  nullptr, HID, S, HID, KVL); // v (f32)

  // ---- rope + concat + rms-norm + head-major relayout ----
  build_q_kernel<<<dim3(S, H), 192, 0, stream>>>(qcf, qrf, Qh);
  build_k_kernel<<<dim3(S, H), 192, 0, stream>>>(kcf, krf, Kh);
  {
    size_t total = (size_t)H * DC * S;
    build_v_kernel<<<(int)((total + 255) / 256), 256, 0, stream>>>(vf, Vt);
  }

  // ---- per-head attention (scores buffer reused across heads) ----
  const float scale = 1.0f / sqrtf((float)DH);
  for (int h = 0; h < H; ++h) {
    const half_t* Qhh = Qh + (size_t)h * S * DH;
    const half_t* Khh = Kh + (size_t)h * S * DH;
    const half_t* Vth = Vt + (size_t)h * DC * S;
    // scores = scale * Q K^T + mask
    gemm(Qhh, DH, Khh, DH, nullptr, mask, S, scale, scf, nullptr, S, S, S, DH);
    // P = softmax(scores)  (f16)
    softmax_row_kernel<<<S, 256, 0, stream>>>(scf, Pmat, S);
    // O[:, h*128 : h*128+128] = P * V   (f16, into [S, HID] layout)
    gemm(Pmat, S, Vth, S, nullptr, nullptr, 0, 1.0f,
         nullptr, Oh + (size_t)h * DC, HID, S, DC, S);
  }

  // ---- final projection: out = O * Wo^T + bo  (f32 -> d_out) ----
  gemm(Oh, HID, Woh, HID, bo, nullptr, 0, 1.0f, out, nullptr, HID, S, HID, HID);
}
